// TemporalInteractionCrossAttention_32942399161113
// MI455X (gfx1250) — matrix operand order repaired and checked by reading, not compiled, not run on hardware
//
#include <hip/hip_runtime.h>
#include <hip/hip_bf16.h>

typedef _Float16 half_t;
typedef __attribute__((ext_vector_type(16))) _Float16 v16h;
typedef __attribute__((ext_vector_type(8)))  _Float16 v8h;
typedef __attribute__((ext_vector_type(4)))  _Float16 v4h;
typedef __attribute__((ext_vector_type(8)))  float    v8f;
typedef __attribute__((ext_vector_type(4)))  float    v4f;

#define B2C 32
#define TT 512
#define DD 512
#define HH 8
#define DHH 64
#define TE 2048
#define NROWS (B2C * TT)      // 16384

// ---------------------------------------------------------------------------
// WMMA helpers (gfx1250 layouts, wave32)
// ---------------------------------------------------------------------------
__device__ __forceinline__ v8f wmma_f16(v16h a, v16h b, v8f c) {
    return __builtin_amdgcn_wmma_f32_16x16x32_f16(
        /*neg_a=*/false, a, /*neg_b=*/false, b,
        /*c_mod=*/(short)0, c, /*reuse_a=*/false, /*reuse_b=*/false);
}

// A fragment (16x32, f16): lane holds row (lane&15); half h -> K = (h<8?h:h+8)+8*hi
// => two contiguous 8-half (16B) chunks at k0 + 8*hi and k0 + 16 + 8*hi.
__device__ __forceinline__ v16h load_a_frag(const half_t* a_tile, int ld, int k0, int lane) {
    int row = lane & 15, hi = (lane >> 4) & 1;
    const half_t* p = a_tile + (size_t)row * ld + k0 + hi * 8;
    v8h lo = *(const v8h*)p;
    v8h hi8 = *(const v8h*)(p + 16);
    v16h a;
#pragma unroll
    for (int i = 0; i < 8; ++i) { a[i] = lo[i]; a[i + 8] = hi8[i]; }
    return a;
}

// B fragment (32x16, f16) from an N-major ("B^T", [N][K] row-major) matrix:
// lane holds column N=(lane&15); half h -> K = (h<8?h:h+8)+8*hi  => two 16B chunks.
__device__ __forceinline__ v16h load_b_frag(const half_t* bt, int ld, int n0, int k0, int lane) {
    int row = lane & 15, hi = (lane >> 4) & 1;
    const half_t* p = bt + (size_t)(n0 + row) * ld + k0 + hi * 8;
    v8h lo = *(const v8h*)p;
    v8h hi8 = *(const v8h*)(p + 16);
    v16h b;
#pragma unroll
    for (int i = 0; i < 8; ++i) { b[i] = lo[i]; b[i + 8] = hi8[i]; }
    return b;
}

__device__ __forceinline__ float silu_f(float x) {
    return x / (1.0f + __expf(-x));
}

// Software-pipelined 16x64 GEMM body: acc += A[16xDD] * BT[n0..n0+64][DD]^T
__device__ __forceinline__ void gemm_16x64(const half_t* arow, const half_t* WT,
                                           int n0, int lane, v8f acc[4]) {
    v16h a = load_a_frag(arow, DD, 0, lane);
    v16h b[4];
#pragma unroll
    for (int j = 0; j < 4; ++j) b[j] = load_b_frag(WT, DD, n0 + 16 * j, 0, lane);

    for (int k0 = 0; k0 < DD - 32; k0 += 32) {
        // issue next k-step's loads before consuming current fragments
        v16h an = load_a_frag(arow, DD, k0 + 32, lane);
        v16h bn[4];
#pragma unroll
        for (int j = 0; j < 4; ++j) bn[j] = load_b_frag(WT, DD, n0 + 16 * j, k0 + 32, lane);
#pragma unroll
        for (int j = 0; j < 4; ++j) acc[j] = wmma_f16(a, b[j], acc[j]);
        a = an;
#pragma unroll
        for (int j = 0; j < 4; ++j) b[j] = bn[j];
    }
#pragma unroll
    for (int j = 0; j < 4; ++j) acc[j] = wmma_f16(a, b[j], acc[j]);
}

// ---------------------------------------------------------------------------
// Kernel: transpose+downconvert weights to f16 W^T   (blockIdx.y selects W)
// ---------------------------------------------------------------------------
__global__ void prep_w_kernel(const float* Wq, const float* Wk, const float* Wv,
                              const float* Wout,
                              half_t* WqT, half_t* WkT, half_t* WvT, half_t* WoutT) {
    int idx = blockIdx.x * 256 + threadIdx.x;   // 0 .. 512*512-1
    int n = idx >> 9, k = idx & 511;
    const float* W; half_t* WT;
    switch (blockIdx.y) {
        case 0:  W = Wq;   WT = WqT;   break;
        case 1:  W = Wk;   WT = WkT;   break;
        case 2:  W = Wv;   WT = WvT;   break;
        default: W = Wout; WT = WoutT; break;
    }
    WT[(size_t)n * DD + k] = (half_t)W[(size_t)k * DD + n];
}

// ---------------------------------------------------------------------------
// Kernel: dual LayerNorm -> f16 (xn from xa, tn from xb).  One wave per row.
// ---------------------------------------------------------------------------
__global__ void ln_to_f16_kernel(const float* x1, const float* x2,
                                 const float* gx, const float* bx,
                                 const float* gt, const float* bt,
                                 half_t* xn, half_t* tn) {
    int lane = threadIdx.x & 31;
    int row  = (blockIdx.x * 256 + threadIdx.x) >> 5;   // 0..16383
    int isT  = blockIdx.y;                              // 0: x-stream, 1: t-stream
    int b2 = row >> 9;
    const float* src;
    if (!isT) src = (b2 < 16) ? (x1 + (size_t)row * DD) : (x2 + (size_t)(row - 8192) * DD);
    else      src = (b2 < 16) ? (x2 + (size_t)row * DD) : (x1 + (size_t)(row - 8192) * DD);
    const float* g = isT ? gt : gx;
    const float* b = isT ? bt : bx;
    half_t* dst = (isT ? tn : xn) + (size_t)row * DD;

    const v4f* s4 = (const v4f*)src;
    v4f vals[4];
    float sum = 0.f, sq = 0.f;
#pragma unroll
    for (int t = 0; t < 4; ++t) {
        v4f v = s4[t * 32 + lane];
        vals[t] = v;
#pragma unroll
        for (int c = 0; c < 4; ++c) { sum += v[c]; sq += v[c] * v[c]; }
    }
#pragma unroll
    for (int m = 1; m <= 16; m <<= 1) {
        sum += __shfl_xor(sum, m, 32);
        sq  += __shfl_xor(sq,  m, 32);
    }
    float mean = sum * (1.0f / DD);
    float var  = sq * (1.0f / DD) - mean * mean;
    float rs   = rsqrtf(var + 1e-5f);

    const v4f* g4 = (const v4f*)g;
    const v4f* b4 = (const v4f*)b;
#pragma unroll
    for (int t = 0; t < 4; ++t) {
        v4f gv = g4[t * 32 + lane];
        v4f bv = b4[t * 32 + lane];
        v4h h;
#pragma unroll
        for (int c = 0; c < 4; ++c)
            h[c] = (half_t)((vals[t][c] - mean) * rs * gv[c] + bv[c]);
        ((v4h*)dst)[t * 32 + lane] = h;
    }
}

// ---------------------------------------------------------------------------
// Kernel: QKV projection GEMM, f16 WMMA.  One wave -> 16x64 tile.
// blockIdx.y: 0=Q (scaled 1/8, row-major), 1=K (row-major), 2=V (transposed)
// ---------------------------------------------------------------------------
__global__ void qkv_gemm_kernel(const half_t* xn, const half_t* tn,
                                const half_t* WqT, const half_t* WkT, const half_t* WvT,
                                const float* bq, const float* bk, const float* bv,
                                half_t* Qh, half_t* Kh, half_t* VTh) {
    int lane = threadIdx.x & 31;
    int task = (blockIdx.x * 256 + threadIdx.x) >> 5;   // 0..8191
    int which = blockIdx.y;
    int mt = task >> 3, nt = task & 7;
    int m0 = mt * 16, n0 = nt * 64;

    const half_t* A  = (which == 0) ? xn : tn;
    const half_t* WT = (which == 0) ? WqT : (which == 1) ? WkT : WvT;
    const half_t* arow = A + (size_t)m0 * DD;

    v8f acc[4];
#pragma unroll
    for (int j = 0; j < 4; ++j)
#pragma unroll
        for (int r = 0; r < 8; ++r) acc[j][r] = 0.f;

    gemm_16x64(arow, WT, n0, lane, acc);

    int rowlo = lane & 15, hi = (lane >> 4) & 1;
    if (which == 0) {
#pragma unroll
        for (int j = 0; j < 4; ++j) {
            int n = n0 + 16 * j + rowlo;
            float bb = bq[n];
#pragma unroll
            for (int r = 0; r < 8; ++r) {
                int row = m0 + r + 8 * hi;
                Qh[(size_t)row * DD + n] = (half_t)((acc[j][r] + bb) * 0.125f);
            }
        }
    } else if (which == 1) {
#pragma unroll
        for (int j = 0; j < 4; ++j) {
            int n = n0 + 16 * j + rowlo;
            float bb = bk[n];
#pragma unroll
            for (int r = 0; r < 8; ++r) {
                int row = m0 + r + 8 * hi;
                Kh[(size_t)row * DD + n] = (half_t)(acc[j][r] + bb);
            }
        }
    } else {
        int b2 = m0 >> 9, t0 = m0 & 511;
#pragma unroll
        for (int j = 0; j < 4; ++j) {
            int n = n0 + 16 * j + rowlo;
            int head = n >> 6, dh = n & 63;
            float bb = bv[n];
            v8h pk;
#pragma unroll
            for (int r = 0; r < 8; ++r) pk[r] = (half_t)(acc[j][r] + bb);
            half_t* dst = VTh + ((size_t)((b2 * HH + head) * DHH + dh)) * TT + t0 + 8 * hi;
            *(v8h*)dst = pk;
        }
    }
}

// ---------------------------------------------------------------------------
// Kernel: silu(emb) precompute
// ---------------------------------------------------------------------------
__global__ void silu_emb_kernel(const float* emb, float* sil) {
    int idx = blockIdx.x * 256 + threadIdx.x;   // 0 .. 32*2048-1
    sil[idx] = silu_f(emb[idx]);
}

// ---------------------------------------------------------------------------
// Kernel: emb MLP  (32 x 2048) @ (2048 x 1024) + bemb -> embout [32][1024]
// ---------------------------------------------------------------------------
__global__ void emb_gemm_kernel(const float* sil, const float* Wemb,
                                const float* bemb, float* embout) {
    int idx = blockIdx.x * 256 + threadIdx.x;   // 0 .. 32*1024-1
    int b2 = idx >> 10, n = idx & 1023;
    const float* s = sil + (size_t)b2 * TE;
    float acc = 0.f;
    for (int k = 0; k < TE; k += 4) {
        acc += s[k + 0] * Wemb[(size_t)(k + 0) * 1024 + n];
        acc += s[k + 1] * Wemb[(size_t)(k + 1) * 1024 + n];
        acc += s[k + 2] * Wemb[(size_t)(k + 2) * 1024 + n];
        acc += s[k + 3] * Wemb[(size_t)(k + 3) * 1024 + n];
    }
    embout[idx] = acc + bemb[n];
}

// ---------------------------------------------------------------------------
// Kernel: flash attention.  One wave -> one (b2, head, 16-row q tile).
// Scores and P.V via f16 WMMA; online softmax; y written f32.
// V fragments are prefetched at the top of each key-tile iteration so they
// overlap the score WMMAs / shuffles / exp chain.
// ---------------------------------------------------------------------------
__global__ void attention_kernel(const half_t* Qh, const half_t* Kh, const half_t* VTh,
                                 const float* mask, float* y) {
    __shared__ half_t plds[8 * 16 * 32];        // 1KB per wave staging for P tiles
    int lane = threadIdx.x & 31;
    int waveId = threadIdx.x >> 5;
    int task = blockIdx.x * 8 + waveId;          // 0..8191
    int qt = task & 31;
    int head = (task >> 5) & 7;
    int b2 = task >> 8;
    int q0 = qt * 16;
    int rowlo = lane & 15, hi = (lane >> 4) & 1;

    const half_t* Qbase  = Qh  + (size_t)(b2 * TT) * DD + head * DHH;
    const half_t* Kbase  = Kh  + (size_t)(b2 * TT) * DD + head * DHH;
    const half_t* VTbase = VTh + (size_t)(b2 * HH + head) * DHH * TT;   // [dh][t]

    v16h a0 = load_a_frag(Qbase + (size_t)q0 * DD, DD, 0,  lane);
    v16h a1 = load_a_frag(Qbase + (size_t)q0 * DD, DD, 32, lane);

    float m_run[8], l_run[8];
    v8f yacc[4];
#pragma unroll
    for (int r = 0; r < 8; ++r) { m_run[r] = -1e30f; l_run[r] = 0.f; }
#pragma unroll
    for (int j = 0; j < 4; ++j)
#pragma unroll
        for (int r = 0; r < 8; ++r) yacc[j][r] = 0.f;

    half_t* pl = plds + waveId * (16 * 32);

    for (int n0 = 0; n0 < TT; n0 += 32) {
        // prefetch V fragments for this key tile (consumed only at the end)
        v16h bvf[4];
#pragma unroll
        for (int j = 0; j < 4; ++j)
            bvf[j] = load_b_frag(VTbase, TT, 16 * j, n0, lane);   // N=dh, K=key

        // prefetch all K fragments for both 16-key subtiles
        v16h kb[4];
#pragma unroll
        for (int st = 0; st < 2; ++st) {
            kb[2 * st + 0] = load_b_frag(Kbase, DD, n0 + st * 16, 0,  lane);
            kb[2 * st + 1] = load_b_frag(Kbase, DD, n0 + st * 16, 32, lane);
        }

        v8f s[2];
#pragma unroll
        for (int st = 0; st < 2; ++st) {
            v8f c;
#pragma unroll
            for (int r = 0; r < 8; ++r) c[r] = 0.f;
            c = wmma_f16(a0, kb[2 * st + 0], c);
            c = wmma_f16(a1, kb[2 * st + 1], c);
            // additive mask: (1 - src_mask) * -1e5
            const float* mp = mask + ((size_t)b2 * TT + q0) * TT + n0 + st * 16 + rowlo;
#pragma unroll
            for (int r = 0; r < 8; ++r)
                c[r] += (1.0f - mp[(size_t)(r + 8 * hi) * TT]) * -100000.0f;
            s[st] = c;
        }
        // per-row max over 32 keys (16 lanes per half-group hold one row's keys)
        float mnew[8];
#pragma unroll
        for (int r = 0; r < 8; ++r) {
            float v = fmaxf(s[0][r], s[1][r]);
#pragma unroll
            for (int m = 1; m <= 8; m <<= 1) v = fmaxf(v, __shfl_xor(v, m, 32));
            mnew[r] = fmaxf(m_run[r], v);
        }
        // rescale running state
#pragma unroll
        for (int r = 0; r < 8; ++r) {
            float corr = __expf(m_run[r] - mnew[r]);
            l_run[r] *= corr;
#pragma unroll
            for (int j = 0; j < 4; ++j) yacc[j][r] *= corr;
            m_run[r] = mnew[r];
        }
        // exponentiate + row sums
#pragma unroll
        for (int r = 0; r < 8; ++r) {
            s[0][r] = __expf(s[0][r] - mnew[r]);
            s[1][r] = __expf(s[1][r] - mnew[r]);
            float rsum = s[0][r] + s[1][r];
#pragma unroll
            for (int m = 1; m <= 8; m <<= 1) rsum += __shfl_xor(rsum, m, 32);
            l_run[r] += rsum;
        }
        // P (C-layout) -> LDS -> A fragment for P.V
        __syncthreads();
#pragma unroll
        for (int st = 0; st < 2; ++st)
#pragma unroll
            for (int r = 0; r < 8; ++r)
                pl[(size_t)(r + 8 * hi) * 32 + st * 16 + rowlo] = (half_t)s[st][r];
        __syncthreads();
        v16h ap = load_a_frag(pl, 32, 0, lane);
#pragma unroll
        for (int j = 0; j < 4; ++j)
            yacc[j] = wmma_f16(ap, bvf[j], yacc[j]);
    }
    // finalize: divide by l, store f32 y [row][D]
    float* yrow = y + ((size_t)(b2 * TT + q0)) * DD + head * DHH;
#pragma unroll
    for (int r = 0; r < 8; ++r) {
        float inv = 1.0f / l_run[r];
#pragma unroll
        for (int j = 0; j < 4; ++j)
            yrow[(size_t)(r + 8 * hi) * DD + 16 * j + rowlo] = yacc[j][r] * inv;
    }
}

// ---------------------------------------------------------------------------
// Kernel: LN(y) * (1+scale) + shift, silu, -> f16 g.  One wave per row.
// ---------------------------------------------------------------------------
__global__ void ln_film_silu_kernel(const float* y, const float* sbg, const float* sbb,
                                    const float* embout, half_t* g) {
    int lane = threadIdx.x & 31;
    int row  = (blockIdx.x * 256 + threadIdx.x) >> 5;
    int b2 = row >> 9;
    const float* src = y + (size_t)row * DD;
    half_t* dst = g + (size_t)row * DD;

    const v4f* s4 = (const v4f*)src;
    v4f vals[4];
    float sum = 0.f, sq = 0.f;
#pragma unroll
    for (int t = 0; t < 4; ++t) {
        v4f v = s4[t * 32 + lane];
        vals[t] = v;
#pragma unroll
        for (int c = 0; c < 4; ++c) { sum += v[c]; sq += v[c] * v[c]; }
    }
#pragma unroll
    for (int m = 1; m <= 16; m <<= 1) {
        sum += __shfl_xor(sum, m, 32);
        sq  += __shfl_xor(sq,  m, 32);
    }
    float mean = sum * (1.0f / DD);
    float var  = sq * (1.0f / DD) - mean * mean;
    float rs   = rsqrtf(var + 1e-5f);

    const v4f* g4 = (const v4f*)sbg;
    const v4f* b4 = (const v4f*)sbb;
    const v4f* sc4 = (const v4f*)(embout + (size_t)b2 * 1024);
    const v4f* sh4 = (const v4f*)(embout + (size_t)b2 * 1024 + 512);
#pragma unroll
    for (int t = 0; t < 4; ++t) {
        v4f gv = g4[t * 32 + lane];
        v4f bv = b4[t * 32 + lane];
        v4f sc = sc4[t * 32 + lane];
        v4f sh = sh4[t * 32 + lane];
        v4h h;
#pragma unroll
        for (int c = 0; c < 4; ++c) {
            float v = (vals[t][c] - mean) * rs * gv[c] + bv[c];
            v = v * (1.0f + sc[c]) + sh[c];
            h[c] = (half_t)silu_f(v);
        }
        ((v4h*)dst)[t * 32 + lane] = h;
    }
}

// ---------------------------------------------------------------------------
// Kernel: out = xa + g @ Wout^T + bout   (f16 WMMA, f32 out).  Wave -> 16x64.
// ---------------------------------------------------------------------------
__global__ void out_gemm_kernel(const half_t* g, const half_t* WoutT,
                                const float* bout, const float* x1, const float* x2,
                                float* out) {
    int lane = threadIdx.x & 31;
    int task = (blockIdx.x * 256 + threadIdx.x) >> 5;   // 0..8191
    int mt = task >> 3, nt = task & 7;
    int m0 = mt * 16, n0 = nt * 64;
    const half_t* arow = g + (size_t)m0 * DD;

    v8f acc[4];
#pragma unroll
    for (int j = 0; j < 4; ++j)
#pragma unroll
        for (int r = 0; r < 8; ++r) acc[j][r] = 0.f;

    gemm_16x64(arow, WoutT, n0, lane, acc);

    int rowlo = lane & 15, hi = (lane >> 4) & 1;
    int b2 = m0 >> 9;
    const float* xabase = (b2 < 16) ? (x1 + (size_t)m0 * DD)
                                    : (x2 + (size_t)(m0 - 8192) * DD);
#pragma unroll
    for (int j = 0; j < 4; ++j) {
        int n = n0 + 16 * j + rowlo;
        float bb = bout[n];
#pragma unroll
        for (int r = 0; r < 8; ++r) {
            int mrow = r + 8 * hi;
            out[(size_t)(m0 + mrow) * DD + n] =
                acc[j][r] + bb + xabase[(size_t)mrow * DD + n];
        }
    }
}

// ---------------------------------------------------------------------------
extern "C" void kernel_launch(void* const* d_in, const int* in_sizes, int n_in,
                              void* d_out, int out_size, void* d_ws, size_t ws_size,
                              hipStream_t stream) {
    const float* x1     = (const float*)d_in[0];
    const float* x2     = (const float*)d_in[1];
    const float* emb    = (const float*)d_in[2];
    const float* mask   = (const float*)d_in[3];
    const float* ln_x_g = (const float*)d_in[4];
    const float* ln_x_b = (const float*)d_in[5];
    const float* ln_t_g = (const float*)d_in[6];
    const float* ln_t_b = (const float*)d_in[7];
    const float* Wq     = (const float*)d_in[8];
    const float* bq     = (const float*)d_in[9];
    const float* Wk     = (const float*)d_in[10];
    const float* bk     = (const float*)d_in[11];
    const float* Wv     = (const float*)d_in[12];
    const float* bv     = (const float*)d_in[13];
    const float* Wemb   = (const float*)d_in[14];
    const float* bemb   = (const float*)d_in[15];
    const float* sb_g   = (const float*)d_in[16];
    const float* sb_b   = (const float*)d_in[17];
    const float* Wout   = (const float*)d_in[18];
    const float* bout   = (const float*)d_in[19];
    float* out = (float*)d_out;

    char* ws = (char*)d_ws;
    size_t off = 0;
    const size_t HROW = (size_t)NROWS * DD * sizeof(half_t);   // 16 MB
    half_t* xn    = (half_t*)(ws + off); off += HROW;
    half_t* tn    = (half_t*)(ws + off); off += HROW;
    half_t* WqT   = (half_t*)(ws + off); off += (size_t)DD * DD * sizeof(half_t);
    half_t* WkT   = (half_t*)(ws + off); off += (size_t)DD * DD * sizeof(half_t);
    half_t* WvT   = (half_t*)(ws + off); off += (size_t)DD * DD * sizeof(half_t);
    half_t* WoutT = (half_t*)(ws + off); off += (size_t)DD * DD * sizeof(half_t);
    half_t* Qh    = (half_t*)(ws + off); off += HROW;
    half_t* Kh    = (half_t*)(ws + off); off += HROW;
    half_t* VTh   = (half_t*)(ws + off); off += HROW;
    float*  yb    = (float*)(ws + off);  off += (size_t)NROWS * DD * sizeof(float);
    float*  sil   = (float*)(ws + off);  off += (size_t)B2C * TE * sizeof(float);
    float*  eo    = (float*)(ws + off);  off += (size_t)B2C * 1024 * sizeof(float);
    half_t* gbuf  = xn;   // xn is dead by the time g is produced

    // 1. weight transpose/downconvert
    prep_w_kernel<<<dim3(1024, 4), 256, 0, stream>>>(Wq, Wk, Wv, Wout,
                                                     WqT, WkT, WvT, WoutT);
    // 2. LayerNorms -> f16
    ln_to_f16_kernel<<<dim3(2048, 2), 256, 0, stream>>>(x1, x2, ln_x_g, ln_x_b,
                                                        ln_t_g, ln_t_b, xn, tn);
    // 3. silu(emb) and emb MLP
    silu_emb_kernel<<<dim3((B2C * TE) / 256), 256, 0, stream>>>(emb, sil);
    emb_gemm_kernel<<<dim3((B2C * 1024) / 256), 256, 0, stream>>>(sil, Wemb, bemb, eo);
    // 4. Q/K/V projections (WMMA)
    qkv_gemm_kernel<<<dim3(1024, 3), 256, 0, stream>>>(xn, tn, WqT, WkT, WvT,
                                                       bq, bk, bv, Qh, Kh, VTh);
    // 5. attention (WMMA + online softmax)
    attention_kernel<<<dim3(1024), 256, 0, stream>>>(Qh, Kh, VTh, mask, yb);
    // 6. LN + FiLM + silu -> f16
    ln_film_silu_kernel<<<dim3(2048), 256, 0, stream>>>(yb, sb_g, sb_b, eo, gbuf);
    // 7. output projection + residual (WMMA)
    out_gemm_kernel<<<dim3(1024), 256, 0, stream>>>(gbuf, WoutT, bout, x1, x2, out);
}
